// MultiHeadAttention_62345745268852
// MI455X (gfx1250) — matrix operand order repaired and checked by reading, not compiled, not run on hardware
//
#include <hip/hip_runtime.h>
#include <stdint.h>

// Problem constants (match reference)
#define BB 2
#define SQL 2048
#define SKL 2048
#define DM 1024
#define NH 16
#define DKH 64

typedef __attribute__((ext_vector_type(16))) __bf16 v16bf;
typedef __attribute__((ext_vector_type(4)))  __bf16 bf16x4;
typedef __attribute__((ext_vector_type(8)))  float  v8f;
typedef __attribute__((ext_vector_type(16))) short  v16s;
typedef __attribute__((ext_vector_type(4)))  unsigned int u32x4;
typedef __attribute__((ext_vector_type(4)))  float  f32x4;

// fp32 -> bf16 through the hardware convert path
__device__ __forceinline__ unsigned short f2bf(float f) {
  __bf16 h = (__bf16)f;
  return __builtin_bit_cast(unsigned short, h);
}

// 16x(16/32) bf16 operand fragment: 8 VGPRs per lane (wave32)
union AF {
  v16s   s;
  u32x4  q[2];
  bf16x4 h[4];
};

__device__ __forceinline__ v8f wmma_bf16(v16s a, v16s b, v8f c) {
  return __builtin_amdgcn_wmma_f32_16x16x32_bf16(
      false, __builtin_bit_cast(v16bf, a),
      false, __builtin_bit_cast(v16bf, b),
      (short)0, c, false, false);
}

// ---------------------------------------------------------------------------
// proj_kernel: Y = X @ W^T.
// Block = 128 threads (4 waves) computing a 64(M) x 64(N) tile.
// Per K-step (32): W sub-tile (64 cols x 32 k) is cooperatively loaded,
// converted to bf16 and staged in LDS (shared by all 4 waves); each wave
// loads one A fragment and reuses it across 4 WMMAs (N tiles).
// mode 0: X=f32, store bf16 Q scaled 1/sqrt(DK), layout [b][h][sq][dk]
// mode 1: X=f32, store bf16 K,                   layout [b][h][sk][dk]
// mode 2: X=f32, store bf16 V transposed,        layout [b][h][dk][sk]
// mode 3: X=bf16(ctx), store f32 directly,       layout [b*sq][D]
// ---------------------------------------------------------------------------
__global__ void mha_proj_kernel(const void* __restrict__ Xv,
                                const float* __restrict__ W,
                                void* __restrict__ Yv, int mode) {
  __shared__ unsigned short Wlds[64 * 40];   // 64 cols x 32 k bf16, padded

  const int tid  = threadIdx.x;
  const int lane = tid & 31;
  const int wave = tid >> 5;
  const int l15  = lane & 15;
  const int hl   = lane >> 4;                // lane half (0/1)

  const int NTB = DM / 64;                   // 16 N-blocks
  const int bm  = blockIdx.x / NTB;          // 64-row block
  const int bn  = blockIdx.x % NTB;          // 64-col block
  const int row = bm * 64 + wave * 16 + l15; // A row for this lane
  const int tnb = bn * 64;

  // cooperative W loader mapping: thread -> (column, k-half)
  const int wcol  = tid >> 1;                // 0..63
  const int whalf = tid & 1;                 // 0..1 (16 k each)
  const float* wsrc = W + (size_t)(tnb + wcol) * DM + whalf * 16;

  const float* Xf = (const float*)Xv;
  const unsigned short* Xb = (const unsigned short*)Xv;

  v8f c[4];
#pragma unroll
  for (int d = 0; d < 4; ++d) c[d] = (v8f){};

  for (int k0 = 0; k0 < DM; k0 += 32) {
    // ---- stage W sub-tile into LDS as bf16 (hardware cvt)
    {
      f32x4 w0 = *(const f32x4*)(wsrc + k0);
      f32x4 w1 = *(const f32x4*)(wsrc + k0 + 4);
      f32x4 w2 = *(const f32x4*)(wsrc + k0 + 8);
      f32x4 w3 = *(const f32x4*)(wsrc + k0 + 12);
      AF t;
      t.h[0] = __builtin_convertvector(w0, bf16x4);
      t.h[1] = __builtin_convertvector(w1, bf16x4);
      t.h[2] = __builtin_convertvector(w2, bf16x4);
      t.h[3] = __builtin_convertvector(w3, bf16x4);
      unsigned short* dst = Wlds + wcol * 40 + whalf * 16;
      *(u32x4*)(dst)     = t.q[0];
      *(u32x4*)(dst + 8) = t.q[1];
      if (k0 + 32 < DM) __builtin_prefetch(wsrc + k0 + 32, 0, 1);
    }
    __syncthreads();

    // ---- A fragment (per-wave rows)
    AF a;
    if (mode == 3) {                         // bf16 input, contiguous K
      const unsigned short* xr = Xb + (size_t)row * DM + k0 + hl * 8;
      a.q[0] = *(const u32x4*)(xr);
      a.q[1] = *(const u32x4*)(xr + 16);
    } else {                                 // fp32 input, convert on load
      const float* xr = Xf + (size_t)row * DM + k0 + hl * 8;
      a.h[0] = __builtin_convertvector(*(const f32x4*)(xr), bf16x4);
      a.h[1] = __builtin_convertvector(*(const f32x4*)(xr + 4), bf16x4);
      a.h[2] = __builtin_convertvector(*(const f32x4*)(xr + 16), bf16x4);
      a.h[3] = __builtin_convertvector(*(const f32x4*)(xr + 20), bf16x4);
    }

    // ---- 4 WMMAs reuse the A fragment across N tiles
#pragma unroll
    for (int d = 0; d < 4; ++d) {
      AF b;
      const unsigned short* wp = Wlds + (d * 16 + l15) * 40 + hl * 8;
      b.q[0] = *(const u32x4*)(wp);
      b.q[1] = *(const u32x4*)(wp + 16);
      c[d] = wmma_bf16(a.s, b.s, c[d]);
    }
    __syncthreads();
  }

  // ---- stores: C fragment VGPR r -> row (r + 8*half), col = lane&15
#pragma unroll
  for (int d = 0; d < 4; ++d) {
#pragma unroll
    for (int r = 0; r < 8; ++r) {
      const int mrow = bm * 64 + wave * 16 + r + 8 * hl;
      const int col  = tnb + d * 16 + l15;
      const float val = c[d][r];
      if (mode == 3) {
        ((float*)Yv)[(size_t)mrow * DM + col] = val;
      } else {
        const int bb = mrow / SQL, ss = mrow % SQL;
        const int hh = col / DKH, dk = col % DKH;
        unsigned short* Y = (unsigned short*)Yv;
        if (mode == 0)
          Y[(((size_t)bb * NH + hh) * SQL + ss) * DKH + dk] = f2bf(val * 0.125f);
        else if (mode == 1)
          Y[(((size_t)bb * NH + hh) * SKL + ss) * DKH + dk] = f2bf(val);
        else
          Y[(((size_t)bb * NH + hh) * DKH + dk) * (size_t)SKL + ss] = f2bf(val);
      }
    }
  }
}

// ---------------------------------------------------------------------------
// attn_kernel: flash-attention style, one wave per 16-query tile per head.
// Online softmax with causal + padding masks; P redistributed via LDS.
// ---------------------------------------------------------------------------
__global__ void mha_attn_kernel(const unsigned short* __restrict__ qb,
                                const unsigned short* __restrict__ kb,
                                const unsigned short* __restrict__ vt,
                                const int* __restrict__ mask,
                                unsigned short* __restrict__ ctxb) {
  __shared__ unsigned short Plds[4][16 * 40];   // per-wave P tile, padded stride

  const int lane = threadIdx.x & 31;
  const int wave = threadIdx.x >> 5;
  const int tile = blockIdx.x * 4 + wave;
  const int QT   = SQL / 16;                    // 128
  const int bh   = tile / QT;
  const int qt   = tile % QT;
  const int b    = bh / NH;
  const int h    = bh % NH;
  const int qbase = qt * 16;
  const int l15  = lane & 15;
  const int hl   = lane >> 4;

  const unsigned short* qhead = qb + (size_t)bh * SQL * DKH;
  const unsigned short* khead = kb + (size_t)bh * SKL * DKH;
  const unsigned short* vhead = vt + (size_t)bh * DKH * SKL;
  const int* mrow = mask + (size_t)b * SKL;

  // Q A-fragments: rows = queries, K = dk (64 -> two 32-chunks)
  AF aq[2];
  {
    const unsigned short* qr = qhead + (size_t)(qbase + l15) * DKH;
#pragma unroll
    for (int kc = 0; kc < 2; ++kc) {
      aq[kc].q[0] = *(const u32x4*)(qr + kc * 32 + hl * 16);
      aq[kc].q[1] = *(const u32x4*)(qr + kc * 32 + 8 + hl * 16);
    }
  }

  float mstat[8], lstat[8];
  v8f acc[4];
#pragma unroll
  for (int r = 0; r < 8; ++r) { mstat[r] = -1e30f; lstat[r] = 0.f; }
#pragma unroll
  for (int d = 0; d < 4; ++d) acc[d] = (v8f){};

  const int kend = qbase + 16;                  // causal bound (exclusive)
  unsigned short* P = &Plds[wave][0];

  for (int kbb = 0; kbb < kend; kbb += 32) {
    // ---- scores S = Q K^T for keys [kbb,kbb+32): two 16x16 f32 fragments
    v8f s0 = {}, s1 = {};
#pragma unroll
    for (int kc = 0; kc < 2; ++kc) {
      AF bk0, bk1;
      const unsigned short* k0p =
          khead + (size_t)(kbb + l15) * DKH + kc * 32 + hl * 16;
      const unsigned short* k1p =
          khead + (size_t)(kbb + 16 + l15) * DKH + kc * 32 + hl * 16;
      bk0.q[0] = *(const u32x4*)(k0p); bk0.q[1] = *(const u32x4*)(k0p + 8);
      bk1.q[0] = *(const u32x4*)(k1p); bk1.q[1] = *(const u32x4*)(k1p + 8);
      s0 = wmma_bf16(aq[kc].s, bk0.s, s0);
      s1 = wmma_bf16(aq[kc].s, bk1.s, s1);
    }
    if (kbb + 32 < kend)
      __builtin_prefetch(khead + (size_t)(kbb + 32 + l15) * DKH, 0, 1);

    // ---- masks (causal + padding), online softmax statistics
    const int j0 = kbb + l15;
    const int j1 = j0 + 16;
    const int v0ok = mrow[j0];
    const int v1ok = mrow[j1];

    float p0[8], p1[8], scl[8];
#pragma unroll
    for (int r = 0; r < 8; ++r) {
      const int qi = qbase + r + 8 * hl;
      float a0 = s0[r], a1 = s1[r];
      if (j0 > qi || !v0ok) a0 = -1e30f;
      if (j1 > qi || !v1ok) a1 = -1e30f;
      float bm = fmaxf(a0, a1);
#pragma unroll
      for (int off = 8; off; off >>= 1)
        bm = fmaxf(bm, __shfl_xor(bm, off, 32));
      const float nm = fmaxf(mstat[r], bm);
      const float sc = __expf(mstat[r] - nm);
      const float e0 = (a0 < -1e29f) ? 0.f : __expf(a0 - nm);
      const float e1 = (a1 < -1e29f) ? 0.f : __expf(a1 - nm);
      float rs = e0 + e1;
#pragma unroll
      for (int off = 8; off; off >>= 1)
        rs += __shfl_xor(rs, off, 32);
      lstat[r] = lstat[r] * sc + rs;
      mstat[r] = nm;
      scl[r] = sc;
      p0[r] = e0;
      p1[r] = e1;
    }

    // rescale existing accumulators (row factor is per-VGPR, uniform in lane group)
#pragma unroll
    for (int d = 0; d < 4; ++d)
#pragma unroll
      for (int r = 0; r < 8; ++r) acc[d][r] *= scl[r];

    // ---- stage P (C-layout, lane=column) into LDS as bf16
#pragma unroll
    for (int r = 0; r < 8; ++r) {
      const int rr = r + 8 * hl;
      P[rr * 40 + l15]      = f2bf(p0[r]);
      P[rr * 40 + 16 + l15] = f2bf(p1[r]);
    }
    asm volatile("s_wait_dscnt 0x0" ::: "memory");

    // ---- reload P as A-fragment (lane=row, K=32 keys)
    AF ap;
    {
      const unsigned short* pr = P + l15 * 40 + hl * 16;
      ap.q[0] = *(const u32x4*)(pr);
      ap.q[1] = *(const u32x4*)(pr + 8);
    }

    // ---- ctx += P @ V  (V transposed: contiguous 16 keys per lane)
#pragma unroll
    for (int d = 0; d < 4; ++d) {
      AF bv;
      const unsigned short* vp =
          vhead + (size_t)(d * 16 + l15) * SKL + kbb + hl * 16;
      bv.q[0] = *(const u32x4*)(vp); bv.q[1] = *(const u32x4*)(vp + 8);
      acc[d] = wmma_bf16(ap.s, bv.s, acc[d]);
    }
  }

  // ---- normalize and store ctx (bf16, [b*sq][D] row-major for O-projection)
#pragma unroll
  for (int r = 0; r < 8; ++r) {
    const int qi = qbase + r + 8 * hl;
    const float inv = lstat[r] > 0.f ? 1.0f / lstat[r] : 0.f;
#pragma unroll
    for (int d = 0; d < 4; ++d) {
      const int col = h * DKH + d * 16 + l15;
      ctxb[((size_t)b * SQL + qi) * DM + col] = f2bf(acc[d][r] * inv);
    }
  }
}

// ---------------------------------------------------------------------------
extern "C" void kernel_launch(void* const* d_in, const int* in_sizes, int n_in,
                              void* d_out, int out_size, void* d_ws, size_t ws_size,
                              hipStream_t stream) {
  const float* query = (const float*)d_in[0];
  const float* key   = (const float*)d_in[1];
  const float* value = (const float*)d_in[2];
  const int*   mask  = (const int*)d_in[3];
  const float* Wq    = (const float*)d_in[4];
  const float* Wk    = (const float*)d_in[5];
  const float* Wv    = (const float*)d_in[6];
  const float* Wo    = (const float*)d_in[7];

  unsigned short* qb  = (unsigned short*)d_ws;
  unsigned short* kb  = qb  + (size_t)BB * NH * SQL * DKH;
  unsigned short* vt  = kb  + (size_t)BB * NH * SKL * DKH;
  unsigned short* ctx = vt  + (size_t)BB * NH * DKH * SKL;

  const int M = BB * SQL;                        // 4096
  const int blocks = (M / 64) * (DM / 64);       // 1024 (64x64 block tiles)
  dim3 blk(128);                                 // 4 waves per block

  mha_proj_kernel<<<dim3(blocks), blk, 0, stream>>>(query, Wq, qb, 0);
  mha_proj_kernel<<<dim3(blocks), blk, 0, stream>>>(key,   Wk, kb, 1);
  mha_proj_kernel<<<dim3(blocks), blk, 0, stream>>>(value, Wv, vt, 2);

  const int atiles = BB * NH * (SQL / 16);       // 4096
  mha_attn_kernel<<<dim3(atiles / 4), blk, 0, stream>>>(qb, kb, vt, mask, ctx);

  mha_proj_kernel<<<dim3(blocks), blk, 0, stream>>>(ctx, Wo, d_out, 3);
}
